// mLSTMBackendTriton_18141941858935
// MI455X (gfx1250) — matrix operand order repaired and checked by reading, not compiled, not run on hardware
//
#include <hip/hip_runtime.h>
#include <cmath>

// Problem constants (fixed by the harness: B=8,H=8,T=4096,D=64,chunk=64)
#define BHN   64
#define TLEN  4096
#define DD    64
#define BT    64
#define NT    64
#define LDM   72          // padded LDS row stride (bf16 elements), multiple of 8
#define LOG2E_F 1.4426950408889634f
#define SCALE_F 0.125f    // D^-0.5

typedef __bf16 v16bf __attribute__((ext_vector_type(16)));
typedef __bf16 v8bf  __attribute__((ext_vector_type(8)));
typedef float  v8f   __attribute__((ext_vector_type(8)));
typedef unsigned int u32x4 __attribute__((ext_vector_type(4)));
typedef int          i32x4 __attribute__((ext_vector_type(4)));
typedef int          i32x8 __attribute__((ext_vector_type(8)));

// ---- CDNA5 bf16 WMMA 16x16x32 wrapper -------------------------------------
__device__ __forceinline__ v8f wmma_bf16(v16bf a, v16bf b, v8f c) {
    // (neg_a, A, neg_b, B, c_mod, C, reuse_a, reuse_b)
    return __builtin_amdgcn_wmma_f32_16x16x32_bf16(false, a, false, b,
                                                   (short)0, c, false, false);
}

// A operand (16x32 bf16): lane m=lane&15, half=lane>>4:
//   elements 0..7  = A[m][k0 + 8*half + 0..7]
//   elements 8..15 = A[m][k0 + 16 + 8*half + 0..7]
__device__ __forceinline__ v16bf fragA(const __bf16* base, int rowblock, int k0, int lane) {
    int m = rowblock * 16 + (lane & 15);
    int half = lane >> 4;
    const __bf16* p = base + m * LDM + k0 + 8 * half;
    v8bf lo = *(const v8bf*)p;
    v8bf hi = *(const v8bf*)(p + 16);
    v16bf f;
#pragma unroll
    for (int e = 0; e < 8; ++e) { f[e] = lo[e]; f[e + 8] = hi[e]; }
    return f;
}

// B operand (32x16 bf16), built from LDS holding B^T row-major (row n = column of B):
//   lane n=lane&15, half=lane>>4: elements 0..7 = B[k0 + 8*half + e][n], 8..15 = B[k0 + 16 + 8*half + e-8][n]
__device__ __forceinline__ v16bf fragB(const __bf16* base, int nblock, int k0, int lane) {
    int n = nblock * 16 + (lane & 15);
    int half = lane >> 4;
    const __bf16* p = base + n * LDM + k0 + 8 * half;
    v8bf lo = *(const v8bf*)p;
    v8bf hi = *(const v8bf*)(p + 16);
    v16bf f;
#pragma unroll
    for (int e = 0; e < 8; ++e) { f[e] = lo[e]; f[e + 8] = hi[e]; }
    return f;
}

// ---------------------------------------------------------------------------
// Kernel 1: per-chunk gate preprocessing.
// grid = BHN*NT blocks, 64 threads. Outputs b (intra-chunk cumsum of log2-sigmoid f),
// g = vecI + f_last - b, and per-chunk f_last / max(g).
// ---------------------------------------------------------------------------
__global__ void gates_kernel(const float* __restrict__ iin, const float* __restrict__ fin,
                             float* __restrict__ bW, float* __restrict__ gW,
                             float* __restrict__ flW, float* __restrict__ gmW) {
    int blk = blockIdx.x;              // bh*NT + nt
    int t   = threadIdx.x;             // 0..63
    int base = blk * BT;               // == bh*TLEN + nt*BT
    __shared__ float a[BT];
    __shared__ float g2[BT];

    float f  = fin[base + t];
    float ls = fminf(f, 0.f) - log1pf(expf(-fabsf(f)));   // log_sigmoid
    a[t] = ls * LOG2E_F;
    __syncthreads();
#pragma unroll
    for (int s = 1; s < BT; s <<= 1) {                    // inclusive scan
        float add = (t >= s) ? a[t - s] : 0.f;
        __syncthreads();
        a[t] += add;
        __syncthreads();
    }
    float b  = a[t];
    float fl = a[BT - 1];
    float vi = iin[base + t] * LOG2E_F;
    float g  = vi + fl - b;
    bW[base + t] = b;
    gW[base + t] = g;
    g2[t] = g;
    __syncthreads();
    if (t == 0) {
        float mx = -1e30f;
        for (int j = 0; j < BT; ++j) mx = fmaxf(mx, g2[j]);
        gmW[blk] = mx;
        flW[blk] = fl;
    }
}

// ---------------------------------------------------------------------------
// Kernel 2: sequential inter-chunk state recurrence.
// grid = BHN*2 (V-dim split in half for occupancy), 256 threads = 8 wave32.
// Keeps C (64 x 32 slice) in WMMA accumulators, updates C = C*dec + kdec^T @ v
// per chunk, and stores the chunk-entry state C^T (bf16), n, m to workspace.
// ---------------------------------------------------------------------------
__global__ void __launch_bounds__(256) phase1_kernel(
        const float* __restrict__ k, const float* __restrict__ v,
        const float* __restrict__ gW, const float* __restrict__ flW,
        const float* __restrict__ gmW,
        __bf16* __restrict__ CT, float* __restrict__ nS, float* __restrict__ mS) {
    int bh  = blockIdx.x >> 1;
    int vsH = blockIdx.x & 1;                 // which half of the V dimension
    int tid = threadIdx.x;
    int lane = tid & 31;
    int w   = tid >> 5;
    int ti  = w >> 1;                         // feature block 0..3 (rows of C)
    int tj  = w & 1;                          // local vdim block 0..1
    int nl  = lane & 15, half = lane >> 4;

    __shared__ __bf16 kT[DD * LDM];           // kdec^T : [feat][key]  (pre-scaled)
    __shared__ __bf16 vT[32 * LDM];           // v^T    : [local vdim][key]
    __shared__ float  nsh[DD];
    __shared__ float  scal[4];                // [0]=m, [1]=m_next, [2]=dec

    if (tid < DD) nsh[tid] = 0.f;
    if (tid == 0) scal[0] = 0.f;
    v8f acc = {};                             // C tile (feat x vdim), fp32
    const float* kb = k + (size_t)bh * TLEN * DD;
    const float* vb = v + (size_t)bh * TLEN * DD;
    __syncthreads();

    for (int nt = 0; nt < NT; ++nt) {
        // --- store chunk-entry state (C^T as bf16, n, m) ---
        {
            size_t cbase = (size_t)(bh * NT + nt) * DD * DD;
#pragma unroll
            for (int r = 0; r < 8; ++r) {
                int feat = ti * 16 + r + 8 * half;
                int vd   = vsH * 32 + tj * 16 + nl;
                CT[cbase + (size_t)vd * DD + feat] = (__bf16)acc[r];
            }
            if (vsH == 0) {
                if (tid < DD) nS[(size_t)(bh * NT + nt) * DD + tid] = nsh[tid];
                if (tid == 0) mS[bh * NT + nt] = scal[0];
            }
        }
        // --- m recurrence (scalar) ---
        if (tid == 0) {
            float m  = scal[0];
            float fl = flW[bh * NT + nt];
            float mn = fmaxf(fl + m, gmW[bh * NT + nt]);
            scal[1] = mn;
            scal[2] = exp2f(fl + m - mn);
        }
        __syncthreads();
        float mn = scal[1], dec = scal[2];
        // --- load k (scaled by exp2(g - m_next)) transposed, v transposed ---
        {
            int key = tid >> 2;
            int c0  = (tid & 3) * 16;
            float gf = exp2f(gW[(size_t)bh * TLEN + nt * BT + key] - mn);
            const float* kr = kb + (size_t)(nt * BT + key) * DD + c0;
#pragma unroll
            for (int cc = 0; cc < 16; ++cc)
                kT[(c0 + cc) * LDM + key] = (__bf16)(kr[cc] * gf);
            int lc0 = (tid & 3) * 8;
            const float* vr = vb + (size_t)(nt * BT + key) * DD + vsH * 32 + lc0;
#pragma unroll
            for (int cc = 0; cc < 8; ++cc)
                vT[(lc0 + cc) * LDM + key] = (__bf16)vr[cc];
        }
        // prefetch next chunk's k/v into cache (global_prefetch_b8)
        if (nt + 1 < NT) {
            __builtin_prefetch(kb + (size_t)(nt + 1) * BT * DD + tid * 16, 0, 1);
            __builtin_prefetch(vb + (size_t)(nt + 1) * BT * DD + tid * 16, 0, 1);
        }
        __syncthreads();
        // --- C = C*dec + kdec^T @ v  (K = 64 keys -> 2 WMMAs) ---
#pragma unroll
        for (int r = 0; r < 8; ++r) acc[r] *= dec;
        {
            v16bf a0 = fragA(kT, ti, 0, lane);
            v16bf b0 = fragB(vT, tj, 0, lane);
            acc = wmma_bf16(a0, b0, acc);
            v16bf a1 = fragA(kT, ti, 32, lane);
            v16bf b1 = fragB(vT, tj, 32, lane);
            acc = wmma_bf16(a1, b1, acc);
        }
        // --- n = n*dec + column-sums of kdec ---
        if (tid < DD) {
            float s = 0.f;
            for (int key = 0; key < BT; ++key) s += (float)kT[tid * LDM + key];
            nsh[tid] = nsh[tid] * dec + s;
        }
        if (tid == 0) scal[0] = scal[1];
        __syncthreads();
    }
}

// ---------------------------------------------------------------------------
// Kernel 3: fully parallel intra-chunk output.
// grid = BHN*NT blocks, 256 threads = 8 wave32, each wave owns two 16x16 tiles.
// h = (rowf * (q@C) + S@v) / denom,  S = (q@k^T) .* scale*exp2(D - m_total).
// The C^T state tile is fetched by the Tensor Data Mover directly into padded
// LDS rows (pad_interval=32 DW, pad_amount=4 DW reproduces the LDM=72 stride).
// ---------------------------------------------------------------------------
__global__ void __launch_bounds__(256) phase2_kernel(
        const float* __restrict__ q, const float* __restrict__ k,
        const float* __restrict__ v, const float* __restrict__ iin,
        const float* __restrict__ bW, const __bf16* __restrict__ CT,
        const float* __restrict__ nS, const float* __restrict__ mS,
        float* __restrict__ out) {
    int blk = blockIdx.x;
    int bh = blk >> 6;                        // NT == 64
    int nt = blk & 63;
    int tid = threadIdx.x;
    int lane = tid & 31;
    int w = tid >> 5;
    int ti = w >> 1;                          // query row block 0..3
    int tj0 = (w & 1) * 2;                    // two adjacent column blocks
    int nl = lane & 15, half = lane >> 4;

    __shared__ __bf16 qs[DD * LDM];           // q       [row][feat]
    __shared__ __bf16 ks[DD * LDM];           // k       [key][feat]   (B of q@k^T)
    __shared__ __bf16 vT2[DD * LDM];          // v^T     [vdim][key]   (B of S@v)
    __shared__ __bf16 cT2[DD * LDM];          // C^T     [vdim][feat]  (B of q@C)
    __shared__ __bf16 sS[BT * LDM];           // S       [row][key]    (A of S@v)
    __shared__ float bL[BT], viL[BT], mtL[BT], rfL[BT], nL[DD], denL[BT];

    size_t tbase = (size_t)bh * TLEN + (size_t)nt * BT;

    // --- TDM: async copy C^T tile (64x64 bf16) -> padded LDS rows -----------
    if (w == 0) {
        const __bf16* gsrc = CT + (size_t)(bh * NT + nt) * DD * DD;
        unsigned long long ga = (unsigned long long)(uintptr_t)gsrc;
        unsigned lds_addr = (unsigned)(uintptr_t)(&cT2[0]);
        u32x4 g0;
        g0[0] = 1u;                                        // count=1 (valid user D#)
        g0[1] = lds_addr;                                  // LDS byte address
        g0[2] = (unsigned)(ga & 0xffffffffu);              // global_addr[31:0]
        g0[3] = ((unsigned)(ga >> 32) & 0x1ffffffu)        // global_addr[56:32]
                | (2u << 30);                              // type = 2 ("image")
        i32x8 g1;
        g1[0] = (int)((1u << 16)                           // data_size = 2 bytes
                    | (1u << 20)                           // pad_enable
                    | (4u << 22)                           // pad_interval: 32 DWORDs
                    | (3u << 25));                         // pad_amount:   4 DWORDs
        g1[1] = (int)(64u << 16);                          // tensor_dim0 = 64 (lo16)
        g1[2] = (int)(64u << 16);                          // dim0 hi=0 | tensor_dim1 lo16 = 64
        g1[3] = (int)(64u << 16);                          // dim1 hi=0 | tile_dim0 = 64
        g1[4] = 64;                                        // tile_dim1 = 64, tile_dim2 = 0
        g1[5] = 64;                                        // tensor_dim0_stride lo32 = 64
        g1[6] = 0;
        g1[7] = 0;
        i32x4 z4 = {};
#if defined(__clang_major__) && __clang_major__ >= 23
        i32x8 z8 = {};
        __builtin_amdgcn_tensor_load_to_lds(g0, g1, z4, z4, z8, 0);
#else
        __builtin_amdgcn_tensor_load_to_lds(g0, g1, z4, z4, 0);
#endif
    }

    if (tid < BT) {
        bL[tid]  = bW[tbase + tid];
        viL[tid] = iin[tbase + tid] * LOG2E_F;
        nL[tid]  = nS[(size_t)(bh * NT + nt) * DD + tid];
    }
    __syncthreads();
    if (tid == 0) {
        // m_total via prefix-max of (vecI - b); rowf = scale*exp2(b + m - m_total)
        float m  = mS[bh * NT + nt];
        float pm = -1e30f;
        for (int i2 = 0; i2 < BT; ++i2) {
            pm = fmaxf(pm, viL[i2] - bL[i2]);
            float mt = bL[i2] + fmaxf(m, pm);
            mtL[i2] = mt;
            rfL[i2] = SCALE_F * exp2f(bL[i2] + m - mt);
        }
    }
    // cooperative global->LDS loads with f32->bf16 convert (v transposed);
    // overlaps with the in-flight TDM transfer of C^T
    {
        int row = tid >> 2;
        int c0  = (tid & 3) * 16;
        const float*  qr = q + (tbase + row) * DD + c0;
        const float*  kr = k + (tbase + row) * DD + c0;
        const float*  vr = v + (tbase + row) * DD + c0;
#pragma unroll
        for (int cc = 0; cc < 16; ++cc) {
            qs[row * LDM + c0 + cc]  = (__bf16)qr[cc];
            ks[row * LDM + c0 + cc]  = (__bf16)kr[cc];
            vT2[(c0 + cc) * LDM + row] = (__bf16)vr[cc];
        }
    }
    if (w == 0) __builtin_amdgcn_s_wait_tensorcnt(0);      // C^T tile landed
    __syncthreads();

    // --- P = q @ k^T (two tiles, shared A fragments) ---
    v8f p0 = {}, p1 = {};
    {
        v16bf a0  = fragA(qs, ti, 0, lane);
        v16bf a1  = fragA(qs, ti, 32, lane);
        v16bf b00 = fragB(ks, tj0, 0, lane);
        v16bf b01 = fragB(ks, tj0, 32, lane);
        v16bf b10 = fragB(ks, tj0 + 1, 0, lane);
        v16bf b11 = fragB(ks, tj0 + 1, 32, lane);
        p0 = wmma_bf16(a0, b00, p0);
        p0 = wmma_bf16(a1, b01, p0);
        p1 = wmma_bf16(a0, b10, p1);
        p1 = wmma_bf16(a1, b11, p1);
    }
    // --- S = P .* scale*exp2(D - m_total), tril mask; spill to LDS as bf16 ---
#pragma unroll
    for (int r = 0; r < 8; ++r) {
        int i = ti * 16 + r + 8 * half;
        {
            int j = tj0 * 16 + nl;
            float s = (j <= i) ? SCALE_F * exp2f(viL[j] - bL[j] + bL[i] - mtL[i]) * p0[r] : 0.f;
            sS[i * LDM + j] = (__bf16)s;
        }
        {
            int j = (tj0 + 1) * 16 + nl;
            float s = (j <= i) ? SCALE_F * exp2f(viL[j] - bL[j] + bL[i] - mtL[i]) * p1[r] : 0.f;
            sS[i * LDM + j] = (__bf16)s;
        }
    }
    __syncthreads();

    // --- u = q @ C ;  h = S @ v ---
    v8f u0 = {}, u1 = {}, h0 = {}, h1 = {};
    {
        v16bf a0  = fragA(qs, ti, 0, lane);
        v16bf a1  = fragA(qs, ti, 32, lane);
        v16bf c00 = fragB(cT2, tj0, 0, lane);
        v16bf c01 = fragB(cT2, tj0, 32, lane);
        v16bf c10 = fragB(cT2, tj0 + 1, 0, lane);
        v16bf c11 = fragB(cT2, tj0 + 1, 32, lane);
        u0 = wmma_bf16(a0, c00, u0);
        u0 = wmma_bf16(a1, c01, u0);
        u1 = wmma_bf16(a0, c10, u1);
        u1 = wmma_bf16(a1, c11, u1);

        v16bf s0  = fragA(sS, ti, 0, lane);
        v16bf s1  = fragA(sS, ti, 32, lane);
        v16bf v00 = fragB(vT2, tj0, 0, lane);
        v16bf v01 = fragB(vT2, tj0, 32, lane);
        v16bf v10 = fragB(vT2, tj0 + 1, 0, lane);
        v16bf v11 = fragB(vT2, tj0 + 1, 32, lane);
        h0 = wmma_bf16(s0, v00, h0);
        h0 = wmma_bf16(s1, v01, h0);
        h1 = wmma_bf16(s0, v10, h1);
        h1 = wmma_bf16(s1, v11, h1);
    }
    // --- denominator per row ---
    if (tid < BT) {
        float rs = 0.f;
        for (int j = 0; j < BT; ++j) rs += (float)sS[tid * LDM + j];
        float qn = 0.f;
        for (int c = 0; c < DD; ++c) qn += (float)qs[tid * LDM + c] * nL[c];
        denL[tid] = fmaxf(fabsf(rfL[tid] * qn + rs), exp2f(-mtL[tid]));
    }
    __syncthreads();
    // --- combine and write out ---
    float* ob = out + tbase * DD;
#pragma unroll
    for (int r = 0; r < 8; ++r) {
        int i = ti * 16 + r + 8 * half;
        float inv = 1.f / denL[i];
        {
            int j = tj0 * 16 + nl;
            ob[(size_t)i * DD + j] = (rfL[i] * u0[r] + h0[r]) * inv;
        }
        {
            int j = (tj0 + 1) * 16 + nl;
            ob[(size_t)i * DD + j] = (rfL[i] * u1[r] + h1[r]) * inv;
        }
    }
}

// ---------------------------------------------------------------------------
extern "C" void kernel_launch(void* const* d_in, const int* in_sizes, int n_in,
                              void* d_out, int out_size, void* d_ws, size_t ws_size,
                              hipStream_t stream) {
    const float* q   = (const float*)d_in[0];
    const float* k   = (const float*)d_in[1];
    const float* v   = (const float*)d_in[2];
    const float* iin = (const float*)d_in[3];
    const float* fin = (const float*)d_in[4];
    float* out = (float*)d_out;

    // workspace carve-up (~37 MB total)
    char* ws = (char*)d_ws;
    size_t off = 0;
    auto carve = [&](size_t bytes) { size_t o = off; off += (bytes + 255) & ~(size_t)255; return o; };
    __bf16* CT  = (__bf16*)(ws + carve((size_t)BHN * NT * DD * DD * sizeof(__bf16)));
    float*  nS  = (float*) (ws + carve((size_t)BHN * NT * DD * sizeof(float)));
    float*  mS  = (float*) (ws + carve((size_t)BHN * NT * sizeof(float)));
    float*  bW  = (float*) (ws + carve((size_t)BHN * TLEN * sizeof(float)));
    float*  gW  = (float*) (ws + carve((size_t)BHN * TLEN * sizeof(float)));
    float*  flW = (float*) (ws + carve((size_t)BHN * NT * sizeof(float)));
    float*  gmW = (float*) (ws + carve((size_t)BHN * NT * sizeof(float)));

    gates_kernel<<<BHN * NT, BT, 0, stream>>>(iin, fin, bW, gW, flW, gmW);
    phase1_kernel<<<BHN * 2, 256, 0, stream>>>(k, v, gW, flW, gmW, CT, nS, mS);
    phase2_kernel<<<BHN * NT, 256, 0, stream>>>(q, k, v, iin, bW, CT, nS, mS, out);
}
